// MultiHeadSelfAttention_59596966199728
// MI455X (gfx1250) — matrix operand order repaired and checked by reading, not compiled, not run on hardware
//
#include <hip/hip_runtime.h>

// MI455X / gfx1250 multi-head self-attention, bf16 WMMA pipeline.
// B=4, S=2048, E=1024, H=16, D=64.

typedef __attribute__((ext_vector_type(16))) __bf16 v16bf;
typedef __attribute__((ext_vector_type(4)))  __bf16 v4bf;
typedef __attribute__((ext_vector_type(8)))  float  v8f;
typedef unsigned int u32;

#define Bsz 4
#define Ssz 2048
#define Esz 1024
#define Hsz 16
#define Dsz 64

// ---- CDNA5 async memory->LDS copies via inline asm (ASYNCcnt-tracked) ----
// VDST = LDS byte offset (low 32 bits of a generic __shared__ pointer = AS3
// offset on amdgcn), VADDR = 64-bit global address, SADDR = off.
static __device__ __forceinline__ void async_cp16(const void* g, void* l) {
    u32 laddr = (u32)(unsigned long long)l;
    asm volatile("global_load_async_to_lds_b128 %0, %1, off"
                 :: "v"(laddr), "v"(g) : "memory");
}

static __device__ __forceinline__ void wait_async_n4() {
    asm volatile("s_wait_asynccnt 0x4" ::: "memory");
}

static __device__ __forceinline__ void wait_async_n0() {
    asm volatile("s_wait_asynccnt 0x0" ::: "memory");
}

union AF { v16bf v; u32 u[8]; };

static __device__ __forceinline__ v8f wmma_bf16(const AF& a, const AF& b, v8f c) {
    return __builtin_amdgcn_wmma_f32_16x16x32_bf16(false, a.v, false, b.v,
                                                   (short)0, c, false, false);
}

// ---------------------------------------------------------------------------
// fp32 -> bf16 conversion (4 elems / thread)
// ---------------------------------------------------------------------------
__global__ void cvt_f32_bf16(const float* __restrict__ in, __bf16* __restrict__ out, int n4) {
    int i = blockIdx.x * 256 + threadIdx.x;
    int stride = gridDim.x * 256;
    for (; i < n4; i += stride) {
        float4 f = ((const float4*)in)[i];
        v4bf o = {(__bf16)f.x, (__bf16)f.y, (__bf16)f.z, (__bf16)f.w};
        ((v4bf*)out)[i] = o;
    }
}

// ---------------------------------------------------------------------------
// GEMM core: C[row, col] = sum_k A[row,k] * Bw[col,k] + bias[col]
// A: [M,1024] bf16 row-major, Bw: [N,1024] bf16 row-major (W, used as B^T).
// 128x128 block tile, 8 waves x (64x32 wave tile), K-step 32,
// double-buffered LDS with async global->LDS copies.
// MODE 0: scatter bf16 into Q/K/V [B,H,S,D].  MODE 1: f32 out [M, 1024].
// ---------------------------------------------------------------------------
template <int MODE>
__global__ __launch_bounds__(256)
void gemm_bf16(const __bf16* __restrict__ A, const __bf16* __restrict__ Bw,
               const float* __restrict__ bias,
               __bf16* __restrict__ Qb, __bf16* __restrict__ Kb, __bf16* __restrict__ Vb,
               float* __restrict__ Out) {
    constexpr int K = 1024;
    __shared__ __bf16 At[2][128 * 32];
    __shared__ __bf16 Bt[2][128 * 32];

    const int tid  = threadIdx.x;
    const int wave = tid >> 5, lane = tid & 31;
    const int half = lane >> 4, l16 = lane & 15;
    const int wm = wave & 1, wn = wave >> 1;
    const int bm = blockIdx.x * 128, bn = blockIdx.y * 128;

    v8f acc[4][2];
    const v8f vz = {0.f,0.f,0.f,0.f,0.f,0.f,0.f,0.f};
#pragma unroll
    for (int i = 0; i < 4; ++i)
#pragma unroll
        for (int j = 0; j < 2; ++j) acc[i][j] = vz;

    const int lrow = tid >> 1;          // 0..127
    const int lk   = (tid & 1) * 16;    // 0 or 16

    const __bf16* ga = A  + (size_t)(bm + lrow) * K + lk;
    const __bf16* gb = Bw + (size_t)(bn + lrow) * K + lk;
    __bf16* la0 = &At[0][lrow * 32 + lk];
    __bf16* lb0 = &Bt[0][lrow * 32 + lk];
    __bf16* la1 = &At[1][lrow * 32 + lk];
    __bf16* lb1 = &Bt[1][lrow * 32 + lk];

    // preload tile 0
    async_cp16(ga, la0);      async_cp16(ga + 8, la0 + 8);
    async_cp16(gb, lb0);      async_cp16(gb + 8, lb0 + 8);

    int buf = 0;
    for (int k0 = 0; k0 < K; k0 += 32) {
        if (k0 + 32 < K) {   // issue next tile into the other buffer
            const __bf16* na = ga + k0 + 32;
            const __bf16* nb = gb + k0 + 32;
            __bf16* ta = buf ? la0 : la1;
            __bf16* tb = buf ? lb0 : lb1;
            async_cp16(na, ta);      async_cp16(na + 8, ta + 8);
            async_cp16(nb, tb);      async_cp16(nb + 8, tb + 8);
            wait_async_n4();         // in-order: current tile's 4 ops done
        } else {
            wait_async_n0();
        }
        __syncthreads();

        const __bf16* Ac = At[buf];
        const __bf16* Bc = Bt[buf];
        AF af[4], bf[2];
#pragma unroll
        for (int i = 0; i < 4; ++i) {
            int r = wm * 64 + i * 16 + l16;
#pragma unroll
            for (int v = 0; v < 8; ++v) {
                int kb = 2 * v + 8 * half + ((v >= 4) ? 8 : 0);   // A 16x32 layout
                af[i].u[v] = *(const u32*)&Ac[r * 32 + kb];
            }
        }
#pragma unroll
        for (int j = 0; j < 2; ++j) {
            int c = wn * 32 + j * 16 + l16;
#pragma unroll
            for (int v = 0; v < 8; ++v) {
                int kb = 2 * v + 8 * half + ((v >= 4) ? 8 : 0);   // B 32x16 layout
                bf[j].u[v] = *(const u32*)&Bc[c * 32 + kb];
            }
        }
#pragma unroll
        for (int i = 0; i < 4; ++i)
#pragma unroll
            for (int j = 0; j < 2; ++j)
                acc[i][j] = wmma_bf16(af[i], bf[j], acc[i][j]);
        __syncthreads();
        buf ^= 1;
    }

#pragma unroll
    for (int i = 0; i < 4; ++i) {
#pragma unroll
        for (int j = 0; j < 2; ++j) {
            int col = bn + wn * 32 + j * 16 + l16;
            float bv = bias[col];
            if (MODE == 0) {
                int t = col >> 10, h = (col >> 6) & 15, d = col & 63;
                __bf16* dst = (t == 0) ? Qb : ((t == 1) ? Kb : Vb);
#pragma unroll
                for (int v = 0; v < 8; ++v) {
                    int row = bm + wm * 64 + i * 16 + v + 8 * half;  // C/D row layout
                    int b = row >> 11, s = row & 2047;
                    dst[((size_t)(b * Hsz + h) * Ssz + s) * Dsz + d] =
                        (__bf16)(acc[i][j][v] + bv);
                }
            } else {
#pragma unroll
                for (int v = 0; v < 8; ++v) {
                    int row = bm + wm * 64 + i * 16 + v + 8 * half;
                    Out[(size_t)row * 1024 + col] = acc[i][j][v] + bv;
                }
            }
        }
    }
}

// ---------------------------------------------------------------------------
// Flash attention: one block per (128 query rows, b*H+h). 8 waves x 16 rows.
// 32-key chunks: scores via 4 WMMA, online softmax (fully unrolled), PV via 4 WMMA.
// ---------------------------------------------------------------------------
__global__ __launch_bounds__(256)
void attn_kernel(const __bf16* __restrict__ Qb, const __bf16* __restrict__ Kb,
                 const __bf16* __restrict__ Vb, const unsigned char* __restrict__ pmask,
                 __bf16* __restrict__ Ob) {
    __shared__ __bf16 Kt[32 * 64];     // [key][d]
    __shared__ __bf16 VtT[64 * 32];    // [d][key] (transposed)
    __shared__ __bf16 Ps[8][16 * 32];  // per-wave P scratch [row][key]

    const int tid  = threadIdx.x;
    const int wave = tid >> 5, lane = tid & 31;
    const int half = lane >> 4, l16 = lane & 15;
    const int bh = blockIdx.y, b = bh >> 4, h = bh & 15;
    const int q0 = blockIdx.x * 128 + wave * 16;
    const float scale = 0.125f;  // D^-0.5

    // Resident Q fragments: 16 rows x 64 (two 16x32 A-fragments)
    AF qf[2];
    {
        const __bf16* Qrow = Qb + ((size_t)bh * Ssz + q0 + l16) * Dsz;
#pragma unroll
        for (int f = 0; f < 2; ++f)
#pragma unroll
            for (int v = 0; v < 8; ++v) {
                int kb = 2 * v + 8 * half + ((v >= 4) ? 8 : 0) + f * 32;
                qf[f].u[v] = *(const u32*)(const void*)(Qrow + kb);
            }
    }

    v8f Oa[4];
    const v8f vz = {0.f,0.f,0.f,0.f,0.f,0.f,0.f,0.f};
#pragma unroll
    for (int nt = 0; nt < 4; ++nt) Oa[nt] = vz;
    float m_i[8], l_i[8];
#pragma unroll
    for (int v = 0; v < 8; ++v) { m_i[v] = -1e30f; l_i[v] = 0.f; }

    const int ldt = tid >> 3;        // key 0..31
    const int ldd = (tid & 7) * 8;   // d base
    const __bf16* Kg = Kb + ((size_t)bh * Ssz + ldt) * Dsz + ldd;
    const __bf16* Vg = Vb + ((size_t)bh * Ssz + ldt) * Dsz + ldd;
    __bf16* Ps_w = Ps[wave];

    for (int c = 0; c < Ssz / 32; ++c) {
        const int kbase = c * 32;
        // prefetch next V chunk while we work (global_prefetch_b8)
        if (c + 1 < Ssz / 32) __builtin_prefetch(Vg + (size_t)(kbase + 32) * Dsz, 0, 0);

        uint4 vv = *(const uint4*)(Vg + (size_t)kbase * Dsz);
        __syncthreads();   // previous chunk fully consumed
        async_cp16(Kg + (size_t)kbase * Dsz, &Kt[ldt * 64 + ldd]);
        const __bf16* vp = (const __bf16*)&vv;
#pragma unroll
        for (int q = 0; q < 8; ++q) VtT[(ldd + q) * 32 + ldt] = vp[q];  // transpose
        wait_async_n0();
        __syncthreads();

        float mb0 = pmask[(size_t)b * Ssz + kbase + l16]      ? 0.f : -1e30f;
        float mb1 = pmask[(size_t)b * Ssz + kbase + 16 + l16] ? 0.f : -1e30f;

        // scores S = Q * K^T  (two 16-column tiles)
        v8f s0 = vz, s1 = vz;
#pragma unroll
        for (int kk = 0; kk < 2; ++kk) {
            AF kf0, kf1;
#pragma unroll
            for (int v = 0; v < 8; ++v) {
                int db = 2 * v + 8 * half + ((v >= 4) ? 8 : 0) + kk * 32;
                kf0.u[v] = *(const u32*)&Kt[l16 * 64 + db];
                kf1.u[v] = *(const u32*)&Kt[(16 + l16) * 64 + db];
            }
            s0 = wmma_bf16(qf[kk], kf0, s0);
            s1 = wmma_bf16(qf[kk], kf1, s1);
        }

        // online softmax (rows: v + 8*half; 16-lane shuffles stay in-half)
#pragma unroll
        for (int v = 0; v < 8; ++v) {
            float x0 = s0[v] * scale + mb0;
            float x1 = s1[v] * scale + mb1;
            float mx = fmaxf(x0, x1);
#pragma unroll
            for (int off = 1; off < 16; off <<= 1) mx = fmaxf(mx, __shfl_xor(mx, off, 32));
            float mn = fmaxf(m_i[v], mx);
            float al = __expf(m_i[v] - mn);
            m_i[v] = mn;
            float p0 = __expf(x0 - mn), p1 = __expf(x1 - mn);
            float rs = p0 + p1;
#pragma unroll
            for (int off = 1; off < 16; off <<= 1) rs += __shfl_xor(rs, off, 32);
            l_i[v] = l_i[v] * al + rs;
#pragma unroll
            for (int nt = 0; nt < 4; ++nt) Oa[nt][v] *= al;
            int pr = (v + 8 * half) * 32;
            Ps_w[pr + l16]      = (__bf16)p0;   // C-layout -> [row][key]
            Ps_w[pr + 16 + l16] = (__bf16)p1;
        }

        // P (16x32) as A-fragment; same-wave LDS is in-order, no barrier needed
        AF pf;
#pragma unroll
        for (int v = 0; v < 8; ++v) {
            int kb = 2 * v + 8 * half + ((v >= 4) ? 8 : 0);
            pf.u[v] = *(const u32*)&Ps_w[l16 * 32 + kb];
        }
        // O += P * V   (four 16-col tiles over D=64)
#pragma unroll
        for (int nt = 0; nt < 4; ++nt) {
            AF vf;
            int dc = nt * 16 + l16;
#pragma unroll
            for (int v = 0; v < 8; ++v) {
                int kb = 2 * v + 8 * half + ((v >= 4) ? 8 : 0);
                vf.u[v] = *(const u32*)&VtT[dc * 32 + kb];
            }
            Oa[nt] = wmma_bf16(pf, vf, Oa[nt]);
        }
    }

#pragma unroll
    for (int nt = 0; nt < 4; ++nt)
#pragma unroll
        for (int v = 0; v < 8; ++v) {
            int row = q0 + v + 8 * half;
            float o = Oa[nt][v] / l_i[v];
            Ob[((size_t)b * Ssz + row) * Esz + h * Dsz + nt * 16 + l16] = (__bf16)o;
        }
}

// ---------------------------------------------------------------------------
extern "C" void kernel_launch(void* const* d_in, const int* in_sizes, int n_in,
                              void* d_out, int out_size, void* d_ws, size_t ws_size,
                              hipStream_t stream) {
    const float* x             = (const float*)d_in[0];
    const unsigned char* pmask = (const unsigned char*)d_in[1];  // jnp bool: 1B/elem
    const float* qkv_w         = (const float*)d_in[2];
    const float* qkv_b         = (const float*)d_in[3];
    const float* proj_w        = (const float*)d_in[4];
    const float* proj_b        = (const float*)d_in[5];
    float* out                 = (float*)d_out;

    char* ws = (char*)d_ws;
    const size_t MB = 1u << 20;
    __bf16* Xb = (__bf16*)(ws);             // 16 MB  [B*S, E]
    __bf16* Wq = (__bf16*)(ws + 16 * MB);   //  6 MB  [3E, E]
    __bf16* Wp = (__bf16*)(ws + 22 * MB);   //  2 MB  [E, E]
    __bf16* Qb = (__bf16*)(ws + 24 * MB);   // 16 MB  [B,H,S,D]
    __bf16* Kb = (__bf16*)(ws + 40 * MB);   // 16 MB
    __bf16* Vb = (__bf16*)(ws + 56 * MB);   // 16 MB
    __bf16* Ob = (__bf16*)(ws + 72 * MB);   // 16 MB  [B*S, E]

    cvt_f32_bf16<<<2048, 256, 0, stream>>>(x,      Xb, (Bsz * Ssz * Esz) / 4);
    cvt_f32_bf16<<<1024, 256, 0, stream>>>(qkv_w,  Wq, (3 * Esz * Esz) / 4);
    cvt_f32_bf16<<<512,  256, 0, stream>>>(proj_w, Wp, (Esz * Esz) / 4);

    // QKV projection: [8192 x 3072] = Xb @ Wq^T + qkv_b, scatter to Q/K/V
    gemm_bf16<0><<<dim3(64, 24), 256, 0, stream>>>(Xb, Wq, qkv_b, Qb, Kb, Vb, nullptr);

    // Attention: 16 row-blocks x 64 (b,h) pairs
    attn_kernel<<<dim3(16, 64), 256, 0, stream>>>(Qb, Kb, Vb, pmask, Ob);

    // Output projection: [8192 x 1024] = Ob @ Wp^T + proj_b (f32 out)
    gemm_bf16<1><<<dim3(64, 8), 256, 0, stream>>>(Ob, Wp, proj_b, nullptr, nullptr, nullptr, out);
}
